// SNNWithSTDP3_13615046328451
// MI455X (gfx1250) — matrix operand order repaired and checked by reading, not compiled, not run on hardware
//
#include <hip/hip_runtime.h>

typedef __attribute__((ext_vector_type(16))) _Float16 v16h;
typedef __attribute__((ext_vector_type(8)))  _Float16 v8h;
typedef __attribute__((ext_vector_type(8)))  float    v8f;

namespace {
constexpr int kT = 50, kB = 256, kI = 1024, kH = 4096, kL = 10;
constexpr float kDtTauMem  = 0.05f;     // DT * TAU_MEM_INV
constexpr float kDtTauSyn  = 0.1f;      // DT * TAU_SYN_INV
constexpr float kVth       = 0.5f;
constexpr float kAPre      = 0.001f;
constexpr float kAPost     = 0.001f;
constexpr float kDtTauPre  = 0.0005f;   // DT * TAU_PRE_INV
constexpr float kDtTauPost = 0.0005f;   // DT * TAU_POST_INV
constexpr float kEtaPlus   = 0.01f;
constexpr float kEtaMinus  = 0.0001f;
constexpr float kTrScale   = 256.0f;    // keep f16 trace copies in normal range
}

// ---- WMMA fragment helpers (wave32, 16x16x32 f16 -> f32) ------------------
// A (MxK=16x32): lane l holds row m=l&15; K halves: (l>>4)*8 + [0..7] (elems 0..7)
//                and 16 + (l>>4)*8 + [0..7] (elems 8..15). Needs row-major, K-contig.
__device__ __forceinline__ v16h ldA(const _Float16* __restrict__ p, int row0,
                                    int k0, int ld) {
  const int l = threadIdx.x & 31;
  const _Float16* r = p + (size_t)(row0 + (l & 15)) * ld + k0 + ((l >> 4) << 3);
  v8h lo = *(const v8h*)r;
  v8h hi = *(const v8h*)(r + 16);
  return __builtin_shufflevector(lo, hi, 0, 1, 2, 3, 4, 5, 6, 7, 8, 9, 10, 11,
                                 12, 13, 14, 15);
}
// B (KxN=32x16): lane l holds col n=l&15; K same as A (table order). Source stored
// as B^T row-major (N rows, K contiguous).
__device__ __forceinline__ v16h ldB(const _Float16* __restrict__ p, int col0,
                                    int k0, int ld) {
  const int l = threadIdx.x & 31;
  const _Float16* r = p + (size_t)(col0 + (l & 15)) * ld + k0 + ((l >> 4) << 3);
  v8h lo = *(const v8h*)r;
  v8h hi = *(const v8h*)(r + 16);
  return __builtin_shufflevector(lo, hi, 0, 1, 2, 3, 4, 5, 6, 7, 8, 9, 10, 11,
                                 12, 13, 14, 15);
}
__device__ __forceinline__ v8f mm(v16h a, v16h b, v8f c) {
  return __builtin_amdgcn_wmma_f32_16x16x32_f16(false, a, false, b, (short)0, c,
                                                false, false);
}

// ---- forward GEMM: in_cur(B,H) = pre16 @ W_in16^T + zprev16 @ W_rec16^T ----
// one 32x32 output tile per wave; 8 waves/block; (256/32)*(4096/32)=1024 tiles
__global__ __launch_bounds__(256) void fwd_gemm_kernel(
    const _Float16* __restrict__ pre16, const _Float16* __restrict__ z16,
    const _Float16* __restrict__ win16, const _Float16* __restrict__ wrec16,
    float* __restrict__ incur) {
  const int wid = threadIdx.x >> 5;
  const int tile = blockIdx.x * 8 + wid;
  const int tn = tile & 127;   // kH/32 = 128 N tiles
  const int tm = tile >> 7;    // kB/32 = 8 M tiles
  const int row0 = tm * 32, col0 = tn * 32;
  v8f a00 = {}, a01 = {}, a10 = {}, a11 = {};
  for (int k = 0; k < kI; k += 32) {
    if (k + 64 < kI) __builtin_prefetch(pre16 + (size_t)row0 * kI + k + 64, 0, 1);
    v16h f0 = ldA(pre16, row0, k, kI);
    v16h f1 = ldA(pre16, row0 + 16, k, kI);
    v16h g0 = ldB(win16, col0, k, kI);
    v16h g1 = ldB(win16, col0 + 16, k, kI);
    a00 = mm(f0, g0, a00); a01 = mm(f0, g1, a01);
    a10 = mm(f1, g0, a10); a11 = mm(f1, g1, a11);
  }
  for (int k = 0; k < kH; k += 32) {
    if (k + 64 < kH) __builtin_prefetch(z16 + (size_t)row0 * kH + k + 64, 0, 1);
    v16h f0 = ldA(z16, row0, k, kH);
    v16h f1 = ldA(z16, row0 + 16, k, kH);
    v16h g0 = ldB(wrec16, col0, k, kH);
    v16h g1 = ldB(wrec16, col0 + 16, k, kH);
    a00 = mm(f0, g0, a00); a01 = mm(f0, g1, a01);
    a10 = mm(f1, g0, a10); a11 = mm(f1, g1, a11);
  }
  const int l = threadIdx.x & 31;
  const int n = l & 15, mb = (l >> 4) << 3;
#pragma unroll
  for (int r = 0; r < 8; ++r) {
    const size_t rm0 = (size_t)(row0 + mb + r) * kH;
    const size_t rm1 = (size_t)(row0 + 16 + mb + r) * kH;
    incur[rm0 + col0 + n]      = a00[r];
    incur[rm0 + col0 + 16 + n] = a01[r];
    incur[rm1 + col0 + n]      = a10[r];
    incur[rm1 + col0 + 16 + n] = a11[r];
  }
}

// ---- STDP weight update: W(H,N) += s1*(A1@B1^T)^? ... K = batch = 256 ------
// delta[m,n] = s1 * sum_k A1[m,k]*B1[n,k] + s2 * sum_k A2[m,k]*B2[n,k]
// also refreshes the f16 shadow copy of W (used by next forward GEMM).
__global__ __launch_bounds__(256) void stdp_upd_kernel(
    float* __restrict__ w, _Float16* __restrict__ w16,
    const _Float16* __restrict__ A1, const _Float16* __restrict__ B1,
    const _Float16* __restrict__ A2, const _Float16* __restrict__ B2, int N,
    float s1, float s2) {
  const int wid = threadIdx.x >> 5;
  const int tile = blockIdx.x * 8 + wid;
  const int ntn = N >> 5;
  const int tn = tile % ntn, tm = tile / ntn;
  const int row0 = tm * 32, col0 = tn * 32;
  v8f p00 = {}, p01 = {}, p10 = {}, p11 = {};
  v8f q00 = {}, q01 = {}, q10 = {}, q11 = {};
#pragma unroll
  for (int k = 0; k < kB; k += 32) {
    v16h f0 = ldA(A1, row0, k, kB);
    v16h f1 = ldA(A1, row0 + 16, k, kB);
    v16h g0 = ldB(B1, col0, k, kB);
    v16h g1 = ldB(B1, col0 + 16, k, kB);
    p00 = mm(f0, g0, p00); p01 = mm(f0, g1, p01);
    p10 = mm(f1, g0, p10); p11 = mm(f1, g1, p11);
    v16h h0 = ldA(A2, row0, k, kB);
    v16h h1 = ldA(A2, row0 + 16, k, kB);
    v16h e0 = ldB(B2, col0, k, kB);
    v16h e1 = ldB(B2, col0 + 16, k, kB);
    q00 = mm(h0, e0, q00); q01 = mm(h0, e1, q01);
    q10 = mm(h1, e0, q10); q11 = mm(h1, e1, q11);
  }
  const int l = threadIdx.x & 31;
  const int n = l & 15, mb = (l >> 4) << 3;
#pragma unroll
  for (int r = 0; r < 8; ++r) {
    const size_t r0 = (size_t)(row0 + mb + r) * N;
    const size_t r1 = (size_t)(row0 + 16 + mb + r) * N;
    const int c0 = col0 + n, c1 = col0 + 16 + n;
    float w00 = w[r0 + c0] + s1 * p00[r] + s2 * q00[r];
    float w01 = w[r0 + c1] + s1 * p01[r] + s2 * q01[r];
    float w10 = w[r1 + c0] + s1 * p10[r] + s2 * q10[r];
    float w11 = w[r1 + c1] + s1 * p11[r] + s2 * q11[r];
    w[r0 + c0] = w00; w16[r0 + c0] = (_Float16)w00;
    w[r0 + c1] = w01; w16[r0 + c1] = (_Float16)w01;
    w[r1 + c0] = w10; w16[r1 + c0] = (_Float16)w10;
    w[r1 + c1] = w11; w16[r1 + c1] = (_Float16)w11;
  }
}

// ---- LIF + post traces over (B,H); LDS 16x16 transposes for f16 copies -----
__global__ __launch_bounds__(256) void lif_stdp_kernel(
    const float* __restrict__ incur, float* __restrict__ v,
    float* __restrict__ isyn, const _Float16* __restrict__ zprev,
    _Float16* __restrict__ zcur, _Float16* __restrict__ zcurT,
    float* __restrict__ tqi, _Float16* __restrict__ tqiT,
    float* __restrict__ tpr, _Float16* __restrict__ tprT,
    float* __restrict__ tqr, _Float16* __restrict__ tqrT,
    float* __restrict__ sc) {
  __shared__ _Float16 tile[16][17];
  const int tx = threadIdx.x, ty = threadIdx.y;
  const int h = blockIdx.x * 16 + tx;
  const int b = blockIdx.y * 16 + ty;
  const size_t idx = (size_t)b * kH + h;
  const size_t tdx = (size_t)(blockIdx.x * 16 + ty) * kB + blockIdx.y * 16 + tx;
  const float vv = v[idx], ii = isyn[idx];
  const float vdec = vv + kDtTauMem * (ii - vv);        // V_LEAK = 0
  const float idec = ii - kDtTauSyn * ii;
  const float z = (vdec - kVth >= 0.0f) ? 1.0f : 0.0f;  // spike()
  v[idx] = (1.0f - z) * vdec;                           // V_RESET = 0
  isyn[idx] = idec + incur[idx];
  zcur[idx] = (_Float16)z;
  const float zp = (float)zprev[idx];
  float tq_i = tqi[idx]; tq_i += kDtTauPost * (kAPost * z - tq_i);  tqi[idx] = tq_i;
  float tp_r = tpr[idx]; tp_r += kDtTauPre  * (kAPre  * zp - tp_r); tpr[idx] = tp_r;
  float tq_r = tqr[idx]; tq_r += kDtTauPost * (kAPost * z - tq_r);  tqr[idx] = tq_r;
  sc[idx] += z;
  tile[ty][tx] = (_Float16)z;                   __syncthreads();
  zcurT[tdx] = tile[tx][ty];                    __syncthreads();
  tile[ty][tx] = (_Float16)(tq_i * kTrScale);   __syncthreads();
  tqiT[tdx] = tile[tx][ty];                     __syncthreads();
  tile[ty][tx] = (_Float16)(tp_r * kTrScale);   __syncthreads();
  tprT[tdx] = tile[tx][ty];                     __syncthreads();
  tile[ty][tx] = (_Float16)(tq_r * kTrScale);   __syncthreads();
  tqrT[tdx] = tile[tx][ty];
}

// ---- input pre-trace over (B,I) --------------------------------------------
__global__ __launch_bounds__(256) void trace_in_kernel(
    const float* __restrict__ xt, float* __restrict__ tpi,
    _Float16* __restrict__ tpiT) {
  __shared__ _Float16 tile[16][17];
  const int tx = threadIdx.x, ty = threadIdx.y;
  const int i = blockIdx.x * 16 + tx;
  const int b = blockIdx.y * 16 + ty;
  const size_t idx = (size_t)b * kI + i;
  float tp = tpi[idx];
  tp += kDtTauPre * (kAPre * xt[idx] - tp);
  tpi[idx] = tp;
  tile[ty][tx] = (_Float16)(tp * kTrScale);
  __syncthreads();
  tpiT[(size_t)(blockIdx.x * 16 + ty) * kB + blockIdx.y * 16 + tx] = tile[tx][ty];
}

// ---- one-time converts -----------------------------------------------------
__global__ __launch_bounds__(256) void convert_x_kernel(
    const float* __restrict__ x, _Float16* __restrict__ x16,
    _Float16* __restrict__ xT16) {
  __shared__ _Float16 tile[16][17];
  const int tx = threadIdx.x, ty = threadIdx.y;
  const int i = blockIdx.x * 16 + tx;
  const int b = blockIdx.y * 16 + ty;
  const size_t base = (size_t)blockIdx.z * kB * kI;
  const float vv = x[base + (size_t)b * kI + i];
  x16[base + (size_t)b * kI + i] = (_Float16)vv;
  tile[ty][tx] = (_Float16)vv;
  __syncthreads();
  xT16[base + (size_t)(blockIdx.x * 16 + ty) * kB + blockIdx.y * 16 + tx] =
      tile[tx][ty];
}

__global__ __launch_bounds__(256) void convert_w_kernel(
    const float* __restrict__ w0, float* __restrict__ w,
    _Float16* __restrict__ w16, int n) {
  const int idx = blockIdx.x * 256 + threadIdx.x;
  if (idx < n) {
    const float vv = w0[idx];
    w[idx] = vv;
    w16[idx] = (_Float16)vv;
  }
}

// ---- outputs ---------------------------------------------------------------
__global__ __launch_bounds__(256) void logits_kernel(
    const float* __restrict__ sc, float* __restrict__ logits) {
  __shared__ float red[256];
  const int b = blockIdx.x, tid = threadIdx.x;
  float acc[kL];
#pragma unroll
  for (int j = 0; j < kL; ++j) acc[j] = 0.0f;
  for (int h = tid; h < kH; h += 256) {
    const float vv = sc[(size_t)b * kH + h];
    int lab = h / 409;            // npd = H // L = 409, clipped to 9
    if (lab > 9) lab = 9;
#pragma unroll
    for (int j = 0; j < kL; ++j) acc[j] += (lab == j) ? vv : 0.0f;
  }
  for (int j = 0; j < kL; ++j) {
    red[tid] = acc[j];
    __syncthreads();
    for (int s = 128; s > 0; s >>= 1) {
      if (tid < s) red[tid] += red[tid + s];
      __syncthreads();
    }
    if (tid == 0) {
      const float cnt = (j == 9) ? 415.0f : 409.0f;
      logits[b * kL + j] = red[0] / (50.0f * cnt);
    }
    __syncthreads();
  }
}

__global__ __launch_bounds__(256) void spikecount_kernel(
    const float* __restrict__ sc, const int* __restrict__ label,
    float* __restrict__ out) {
  const int h = blockIdx.x * 256 + threadIdx.x;
  float acc[kL];
#pragma unroll
  for (int j = 0; j < kL; ++j) acc[j] = 0.0f;
  for (int b = 0; b < kB; ++b) {
    const int lab = label[b];
    const float vv = sc[(size_t)b * kH + h];
#pragma unroll
    for (int j = 0; j < kL; ++j) acc[j] += (lab == j) ? vv : 0.0f;
  }
#pragma unroll
  for (int j = 0; j < kL; ++j) out[(size_t)j * kH + h] = acc[j];
}

__global__ void labelcount_kernel(const int* __restrict__ label,
                                  float* __restrict__ out) {
  __shared__ int bins[kL];
  const int tid = threadIdx.x;
  if (tid < kL) bins[tid] = 0;
  __syncthreads();
  atomicAdd(&bins[label[tid]], 1);   // integer LDS atomics: deterministic
  __syncthreads();
  if (tid < kL) out[tid] = 50.0f * (float)bins[tid];
}

__global__ __launch_bounds__(256) void reduce1_kernel(
    const float* __restrict__ sc, float* __restrict__ part) {
  __shared__ float red[256];
  const int tid = threadIdx.x;
  const size_t n = (size_t)kB * kH;
  float a = 0.0f;
  for (size_t idx = (size_t)blockIdx.x * 256 + tid; idx < n; idx += 256 * 256)
    a += sc[idx];
  red[tid] = a;
  __syncthreads();
  for (int s = 128; s > 0; s >>= 1) {
    if (tid < s) red[tid] += red[tid + s];
    __syncthreads();
  }
  if (tid == 0) part[blockIdx.x] = red[0];
}

__global__ void reduce2_kernel(const float* __restrict__ part,
                               float* __restrict__ total) {
  __shared__ float red[256];
  const int tid = threadIdx.x;
  red[tid] = part[tid];
  __syncthreads();
  for (int s = 128; s > 0; s >>= 1) {
    if (tid < s) red[tid] += red[tid + s];
    __syncthreads();
  }
  if (tid == 0) total[0] = red[0];
}

extern "C" void kernel_launch(void* const* d_in, const int* in_sizes, int n_in,
                              void* d_out, int out_size, void* d_ws,
                              size_t ws_size, hipStream_t stream) {
  (void)in_sizes; (void)n_in; (void)out_size; (void)ws_size;
  const float* x      = (const float*)d_in[0];
  const int*   label  = (const int*)d_in[1];
  const float* w_in0  = (const float*)d_in[2];
  const float* w_rec0 = (const float*)d_in[3];

  float* out      = (float*)d_out;
  float* logits   = out;                             // (256,10)
  float* total    = out + 2560;                      // scalar
  float* w_in     = out + 2561;                      // (4096,1024) fp32 master
  float* w_rec    = w_in + (size_t)kH * kI;          // (4096,4096)
  float* spikecnt = w_rec + (size_t)kH * kH;         // (10,4096)
  float* labelcnt = spikecnt + (size_t)kL * kH;      // (10,)

  char* p = (char*)d_ws;
  auto carve = [&](size_t bytes) -> void* {
    void* r = (void*)p;
    p += (bytes + 255) & ~(size_t)255;
    return r;
  };
  _Float16* x16    = (_Float16*)carve((size_t)kT * kB * kI * 2);  // (T,B,I)
  _Float16* xT16   = (_Float16*)carve((size_t)kT * kB * kI * 2);  // (T,I,B)
  _Float16* win16  = (_Float16*)carve((size_t)kH * kI * 2);
  _Float16* wrec16 = (_Float16*)carve((size_t)kH * kH * 2);
  float*    incur  = (float*)carve((size_t)kB * kH * 4);
  _Float16* tpiT   = (_Float16*)carve((size_t)kI * kB * 2);       // (I,B) x256
  _Float16* tqiT   = (_Float16*)carve((size_t)kH * kB * 2);       // (H,B) x256
  _Float16* tprT   = (_Float16*)carve((size_t)kH * kB * 2);
  _Float16* tqrT   = (_Float16*)carve((size_t)kH * kB * 2);
  float*    part   = (float*)carve(256 * 4);
  // zero-initialized state (kept contiguous for a single memset)
  char* zbase = p;
  float*    v     = (float*)carve((size_t)kB * kH * 4);
  float*    isyn  = (float*)carve((size_t)kB * kH * 4);
  float*    tpi   = (float*)carve((size_t)kB * kI * 4);
  float*    tqi   = (float*)carve((size_t)kB * kH * 4);
  float*    tpr   = (float*)carve((size_t)kB * kH * 4);
  float*    tqr   = (float*)carve((size_t)kB * kH * 4);
  float*    sc    = (float*)carve((size_t)kB * kH * 4);
  _Float16* z16a  = (_Float16*)carve((size_t)kB * kH * 2);
  _Float16* z16b  = (_Float16*)carve((size_t)kB * kH * 2);
  _Float16* zT16a = (_Float16*)carve((size_t)kB * kH * 2);
  _Float16* zT16b = (_Float16*)carve((size_t)kB * kH * 2);
  hipMemsetAsync(zbase, 0, (size_t)(p - zbase), stream);
  hipMemsetAsync(total, 0, sizeof(float), stream);

  convert_w_kernel<<<(kH * kI + 255) / 256, 256, 0, stream>>>(w_in0, w_in,
                                                              win16, kH * kI);
  convert_w_kernel<<<(kH * kH + 255) / 256, 256, 0, stream>>>(w_rec0, w_rec,
                                                              wrec16, kH * kH);
  convert_x_kernel<<<dim3(kI / 16, kB / 16, kT), dim3(16, 16), 0, stream>>>(
      x, x16, xT16);

  const float s1 = kEtaPlus / kTrScale;    // trace f16 copies carry x256
  const float s2 = -kEtaMinus / kTrScale;
  for (int t = 0; t < kT; ++t) {
    _Float16* zp  = (t & 1) ? z16b : z16a;
    _Float16* zc  = (t & 1) ? z16a : z16b;
    _Float16* zTp = (t & 1) ? zT16b : zT16a;
    _Float16* zTc = (t & 1) ? zT16a : zT16b;
    const _Float16* pre16  = x16 + (size_t)t * kB * kI;
    const _Float16* preT16 = xT16 + (size_t)t * kB * kI;

    fwd_gemm_kernel<<<128, 256, 0, stream>>>(pre16, zp, win16, wrec16, incur);
    lif_stdp_kernel<<<dim3(kH / 16, kB / 16), dim3(16, 16), 0, stream>>>(
        incur, v, isyn, zp, zc, zTc, tqi, tqiT, tpr, tprT, tqr, tqrT, sc);
    trace_in_kernel<<<dim3(kI / 16, kB / 16), dim3(16, 16), 0, stream>>>(
        x + (size_t)t * kB * kI, tpi, tpiT);
    // W_in += eta+ * z_new^T @ tp_i  -  eta- * tq_i^T @ pre
    stdp_upd_kernel<<<(kH / 32) * (kI / 32) / 8, 256, 0, stream>>>(
        w_in, win16, zTc, tpiT, tqiT, preT16, kI, s1, s2);
    // W_rec += eta+ * z_new^T @ tp_r -  eta- * tq_r^T @ z_prev
    stdp_upd_kernel<<<(kH / 32) * (kH / 32) / 8, 256, 0, stream>>>(
        w_rec, wrec16, zTc, tprT, tqrT, zTp, kH, s1, s2);
  }

  logits_kernel<<<kB, 256, 0, stream>>>(sc, logits);
  spikecount_kernel<<<kH / 256, 256, 0, stream>>>(sc, label, spikecnt);
  labelcount_kernel<<<1, kB, 0, stream>>>(label, labelcnt);
  reduce1_kernel<<<256, 256, 0, stream>>>(sc, part);
  reduce2_kernel<<<1, 256, 0, stream>>>(part, total);
}
